// ScalarQuantizedXWLinear_72413148610815
// MI455X (gfx1250) — 8-bit matrix operand order repaired and checked by reading, compiled, not run on hardware
//
#include <hip/hip_runtime.h>

typedef __attribute__((ext_vector_type(8))) int v8i;

#define D_IN   4096
#define D_OUT  4096
#define TOKENS 8192
#define QMAX   127.0f
#define EPSQ   1e-8f

#define KSTEPS (D_IN / 64)
#define MT16   (TOKENS / 16)
#define NT16   (D_OUT / 16)

#define A_DWORDS ((size_t)MT16 * KSTEPS * 32 * 8)
#define B_DWORDS ((size_t)NT16 * KSTEPS * 32 * 8)

__device__ __forceinline__ unsigned pack4(float a, float b, float c, float d, float inv) {
  int q0 = (int)fminf(fmaxf(rintf(a * inv), -QMAX), QMAX);
  int q1 = (int)fminf(fmaxf(rintf(b * inv), -QMAX), QMAX);
  int q2 = (int)fminf(fmaxf(rintf(c * inv), -QMAX), QMAX);
  int q3 = (int)fminf(fmaxf(rintf(d * inv), -QMAX), QMAX);
  return (unsigned)(q0 & 255) | ((unsigned)(q1 & 255) << 8) |
         ((unsigned)(q2 & 255) << 16) | ((unsigned)(q3 & 255) << 24);
}

__global__ __launch_bounds__(256) void amax_kernel(const float* __restrict__ w, int n,
                                                   int* __restrict__ amax_bits) {
  __shared__ float red[256];
  float m = 0.0f;
  for (int i = blockIdx.x * blockDim.x + threadIdx.x; i < n; i += gridDim.x * blockDim.x)
    m = fmaxf(m, fabsf(w[i]));
  red[threadIdx.x] = m;
  __syncthreads();
  for (int s = 128; s > 0; s >>= 1) {
    if (threadIdx.x < (unsigned)s)
      red[threadIdx.x] = fmaxf(red[threadIdx.x], red[threadIdx.x + s]);
    __syncthreads();
  }
  if (threadIdx.x == 0) atomicMax(amax_bits, __float_as_int(red[0]));
}

__global__ __launch_bounds__(256) void quantW_pack(const float* __restrict__ W,
                                                   const int* __restrict__ amax_bits,
                                                   unsigned* __restrict__ pB) {
  size_t idx = (size_t)blockIdx.x * blockDim.x + threadIdx.x;
  if (idx >= B_DWORDS) return;
  const float amax = __int_as_float(*amax_bits);
  const float inv  = QMAX / fmaxf(amax, 1e-30f);

  const int d    = (int)(idx & 7);
  const int lane = (int)((idx >> 3) & 31);
  const int ks   = (int)((idx >> 8) & (KSTEPS - 1));
  const int nt   = (int)(idx >> 14);

  const int col = nt * 16 + (lane & 15);
  const int kb  = ks * 64 + (d >> 2) * 32 + ((lane >> 4) & 1) * 16 + (d & 3) * 4;

  const float w0 = W[(size_t)(kb + 0) * D_OUT + col];
  const float w1 = W[(size_t)(kb + 1) * D_OUT + col];
  const float w2 = W[(size_t)(kb + 2) * D_OUT + col];
  const float w3 = W[(size_t)(kb + 3) * D_OUT + col];
  pB[idx] = pack4(w0, w1, w2, w3, inv);
}

__global__ __launch_bounds__(256) void quantX_pack(const float* __restrict__ X,
                                                   const float* __restrict__ x_scale,
                                                   unsigned* __restrict__ pA) {
  size_t idx = (size_t)blockIdx.x * blockDim.x + threadIdx.x;
  if (idx >= A_DWORDS) return;
  const float inv = QMAX / fmaxf(x_scale[0], EPSQ);

  const int d    = (int)(idx & 7);
  const int lane = (int)((idx >> 3) & 31);
  const int ks   = (int)((idx >> 8) & (KSTEPS - 1));
  const int mt   = (int)(idx >> 14);

  const int row = mt * 16 + (lane & 15);
  const int kb  = ks * 64 + (d >> 2) * 32 + ((lane >> 4) & 1) * 16 + (d & 3) * 4;

  const float* src = X + (size_t)row * D_IN + kb;
  pA[idx] = pack4(src[0], src[1], src[2], src[3], inv);
}

__global__ __launch_bounds__(256) void gemm_i8_wmma(const unsigned* __restrict__ pA,
                                                    const unsigned* __restrict__ pB,
                                                    const float* __restrict__ bias,
                                                    const int* __restrict__ amax_bits,
                                                    const float* __restrict__ x_scale,
                                                    float* __restrict__ out) {
  const int lane = threadIdx.x & 31;
  const int wv   = threadIdx.x >> 5;
  const int wm   = wv & 1;
  const int wn   = wv >> 1;

  const int mt0 = blockIdx.y * 8 + wm * 4;
  const int nt0 = blockIdx.x * 8 + wn * 2;

  v8i acc[4][2] = {};

  const unsigned* aBase = pA + ((size_t)mt0 * KSTEPS * 32 + lane) * 8;
  const unsigned* bBase = pB + ((size_t)nt0 * KSTEPS * 32 + lane) * 8;

  #pragma unroll 2
  for (int ks = 0; ks < KSTEPS; ++ks) {
    v8i a[4], b[2];
    #pragma unroll
    for (int i = 0; i < 4; ++i)
      a[i] = *(const v8i*)(aBase + (size_t)i * KSTEPS * 256 + (size_t)ks * 256);
    #pragma unroll
    for (int j = 0; j < 2; ++j)
      b[j] = *(const v8i*)(bBase + (size_t)j * KSTEPS * 256 + (size_t)ks * 256);

    #pragma unroll
    for (int i = 0; i < 4; ++i) {
      #pragma unroll
      for (int j = 0; j < 2; ++j) {
        acc[i][j] = __builtin_amdgcn_wmma_i32_16x16x64_iu8(
            true, a[i], true, b[j], acc[i][j], false, false);
      }
    }
  }

  const float amax   = __int_as_float(*amax_bits);
  const float wscale = amax / QMAX;
  const float xstep  = fmaxf(x_scale[0], EPSQ) / QMAX;
  const float s      = wscale * xstep;

  const int colLo = lane & 15;
  const int rhalf = (lane >> 4) * 8;

  #pragma unroll
  for (int j = 0; j < 2; ++j) {
    const int col = (nt0 + j) * 16 + colLo;
    const float bj = bias[col];
    #pragma unroll
    for (int i = 0; i < 4; ++i) {
      const int row0 = (mt0 + i) * 16 + rhalf;
      #pragma unroll
      for (int r = 0; r < 8; ++r) {
        out[(size_t)(row0 + r) * D_OUT + col] = (float)acc[i][j][r] * s + bj;
      }
    }
  }
}

extern "C" void kernel_launch(void* const* d_in, const int* in_sizes, int n_in,
                              void* d_out, int out_size, void* d_ws, size_t ws_size,
                              hipStream_t stream) {
  const float* X       = (const float*)d_in[0];
  const float* W       = (const float*)d_in[1];
  const float* x_scale = (const float*)d_in[2];
  const float* bias    = (const float*)d_in[3];
  float* out           = (float*)d_out;

  int*      amax_bits = (int*)d_ws;
  unsigned* pA        = (unsigned*)((char*)d_ws + 256);
  unsigned* pB        = (unsigned*)((char*)d_ws + 256 + A_DWORDS * sizeof(unsigned));

  hipMemsetAsync(d_ws, 0, sizeof(int), stream);

  amax_kernel<<<1024, 256, 0, stream>>>(W, D_IN * D_OUT, amax_bits);
  quantW_pack<<<(unsigned)(B_DWORDS / 256), 256, 0, stream>>>(W, amax_bits, pB);
  quantX_pack<<<(unsigned)(A_DWORDS / 256), 256, 0, stream>>>(X, x_scale, pA);

  dim3 grid(D_OUT / 128, TOKENS / 128);
  gemm_i8_wmma<<<grid, 256, 0, stream>>>(pA, pB, bias, amax_bits, x_scale, out);
}
